// GatedDeltaNet_29420525977744
// MI455X (gfx1250) — matrix operand order repaired and checked by reading, not compiled, not run on hardware
//
#include <hip/hip_runtime.h>
#include <hip/hip_bf16.h>
#include <stdint.h>

// ---------------- problem constants ----------------
#define B_  4
#define T_  1024
#define HID_ 1024
#define H_  8
#define DK_ 128
#define DV_ 128
#define KCONV_ 4
#define KEY_DIM_  (H_ * DK_)          // 1024
#define VAL_DIM_  (H_ * DV_)          // 1024
#define CONV_DIM_ (2 * KEY_DIM_ + VAL_DIM_)  // 3072
#define ROWS_ (B_ * T_)               // 4096

// ---------------- WMMA types ----------------
typedef __attribute__((ext_vector_type(16))) __bf16 v16bf;
typedef __attribute__((ext_vector_type(8)))  float  v8f;

// =====================================================================
// GEMM: C[M,N] = X[M,K] * W[N,K]^T   (fp32 in/out, bf16 WMMA, f32 accum)
// Block: 256 threads (8 wave32), computes 128x128 tile of C.
// Wave grid 4(M) x 2(N): each wave owns 32 rows x 64 cols = 2x4 WMMA tiles.
// =====================================================================
#define TILE_M 128
#define TILE_N 128
#define TILE_K 32
#define ASTRIDE 40   // LDS row stride in bf16 elements (padded, conflict-free)

union FragBF { v16bf v; uint32_t u[8]; };

__global__ __launch_bounds__(256) void gemm_xwT_bf16(
    const float* __restrict__ X, const float* __restrict__ W,
    float* __restrict__ C, int M, int N, int K)
{
    __shared__ __bf16 As[TILE_M * ASTRIDE];
    __shared__ __bf16 Ws[TILE_N * ASTRIDE];

    const int tid  = threadIdx.x;
    const int wid  = tid >> 5;
    const int lane = tid & 31;
    const int wr   = wid >> 1;     // 0..3 : M direction
    const int wc   = wid & 1;      // 0..1 : N direction
    const int blockM = blockIdx.y * TILE_M;
    const int blockN = blockIdx.x * TILE_N;

    const int h  = lane >> 4;      // half-wave select
    const int ln = lane & 15;

    v8f acc[2][4];
    const v8f vzero = {0.f,0.f,0.f,0.f,0.f,0.f,0.f,0.f};
#pragma unroll
    for (int i = 0; i < 2; ++i)
#pragma unroll
        for (int j = 0; j < 4; ++j) acc[i][j] = vzero;

    // tile loader: thread loads 16 consecutive K-elements of one row
    const int lr = tid >> 1;
    const int lc = (tid & 1) * 16;

    for (int kb = 0; kb < K; kb += TILE_K) {
        {
            const float* src = X + (size_t)(blockM + lr) * K + kb + lc;
            __bf16* dst = As + lr * ASTRIDE + lc;
#pragma unroll
            for (int i = 0; i < 16; ++i) dst[i] = (__bf16)src[i];
        }
        {
            const float* src = W + (size_t)(blockN + lr) * K + kb + lc;
            __bf16* dst = Ws + lr * ASTRIDE + lc;
#pragma unroll
            for (int i = 0; i < 16; ++i) dst[i] = (__bf16)src[i];
        }
        __syncthreads();

        // A fragments per ISA 16-bit A 16x32 layout:
        // VGPR v<4 : K = 2v,2v+1   (+8 for lanes 16-31)
        // VGPR v>=4: K = 16+2(v-4) (+8 for lanes 16-31)
        FragBF afrag[2];
#pragma unroll
        for (int mt = 0; mt < 2; ++mt) {
            const int m0 = wr * 32 + mt * 16 + ln;
            const uint32_t* row = (const uint32_t*)(As + m0 * ASTRIDE);
#pragma unroll
            for (int v = 0; v < 8; ++v) {
                const int k0 = (v < 4 ? 2 * v : 16 + 2 * (v - 4)) + 8 * h;
                afrag[mt].u[v] = row[k0 >> 1];
            }
        }

#pragma unroll
        for (int nt = 0; nt < 4; ++nt) {
            // B fragment, same K order as A: VGPR v<4: K=2v,2v+1; v>=4: K=16+2(v-4)
            // (+8 for lanes 16-31), packed 2 per VGPR.
            FragBF bfrag;
            const int n0 = wc * 64 + nt * 16 + ln;
            const uint32_t* row = (const uint32_t*)(Ws + n0 * ASTRIDE);
#pragma unroll
            for (int v = 0; v < 8; ++v) {
                const int k0 = (v < 4 ? 2 * v : 16 + 2 * (v - 4)) + 8 * h;
                bfrag.u[v] = row[k0 >> 1];
            }
#pragma unroll
            for (int mt = 0; mt < 2; ++mt) {
                acc[mt][nt] = __builtin_amdgcn_wmma_f32_16x16x32_bf16(
                    false, afrag[mt].v, false, bfrag.v,
                    (short)0, acc[mt][nt], false, false);
            }
        }
        __syncthreads();
    }

    // C layout: VGPR r -> row r (lanes 0-15) / row r+8 (lanes 16-31), col = lane&15
#pragma unroll
    for (int mt = 0; mt < 2; ++mt) {
#pragma unroll
        for (int nt = 0; nt < 4; ++nt) {
            const int col = blockN + wc * 64 + nt * 16 + ln;
            float* p = C + (size_t)(blockM + wr * 32 + mt * 16 + 8 * h) * N + col;
#pragma unroll
            for (int r = 0; r < 8; ++r) {
                *p = acc[mt][nt][r];
                p += N;
            }
        }
    }
}

// =====================================================================
// Causal depthwise conv1d over T (K=4), cross-correlation w/ left pad 3
// =====================================================================
__global__ void conv_kernel(const float* __restrict__ mixed,
                            const float* __restrict__ convw,
                            float* __restrict__ out, int total)
{
    int idx = blockIdx.x * blockDim.x + threadIdx.x;
    if (idx >= total) return;
    int c = idx % CONV_DIM_;
    int t = (idx / CONV_DIM_) % T_;
    int b = idx / (CONV_DIM_ * T_);
    const float* w = convw + c * KCONV_;
    float acc = 0.f;
#pragma unroll
    for (int j = 0; j < KCONV_; ++j) {
        int tt = t + j - (KCONV_ - 1);
        if (tt >= 0)
            acc += w[j] * mixed[((size_t)b * T_ + tt) * CONV_DIM_ + c];
    }
    out[idx] = acc;
}

// =====================================================================
// Small projections bb/aa: one wave32 per output scalar
// =====================================================================
__global__ __launch_bounds__(256) void ba_proj_kernel(
    const float* __restrict__ hs, const float* __restrict__ b_w,
    const float* __restrict__ a_w, float* __restrict__ bb,
    float* __restrict__ aa)
{
    const int wave = (blockIdx.x * blockDim.x + threadIdx.x) >> 5;
    const int lane = threadIdx.x & 31;
    const int total = ROWS_ * H_ * 2;
    if (wave >= total) return;
    const int which = wave / (ROWS_ * H_);
    const int rem   = wave % (ROWS_ * H_);
    const int row = rem / H_;
    const int hh  = rem % H_;
    const float* w = (which ? a_w : b_w) + (size_t)hh * HID_;
    const float* x = hs + (size_t)row * HID_;
    float s = 0.f;
    for (int k = lane; k < HID_; k += 32) s += x[k] * w[k];
#pragma unroll
    for (int off = 16; off; off >>= 1) s += __shfl_xor(s, off, 32);
    if (lane == 0) (which ? aa : bb)[rem] = s;
}

// =====================================================================
// alpha / beta gates
// =====================================================================
__global__ void gate_kernel(const float* __restrict__ bb,
                            const float* __restrict__ aa,
                            const float* __restrict__ A_log,
                            const float* __restrict__ dt_bias,
                            float* __restrict__ alpha,
                            float* __restrict__ beta, int total)
{
    int i = blockIdx.x * blockDim.x + threadIdx.x;
    if (i >= total) return;
    int hh = i % H_;
    float x  = aa[i] + dt_bias[hh];
    float sp = (x > 20.f) ? x : log1pf(__expf(x));
    float g  = -__expf(A_log[hh]) * sp;
    alpha[i] = __expf(g);
    beta[i]  = 1.f / (1.f + __expf(-bb[i]));
}

// =====================================================================
// Gated delta-rule recurrence. One block per (b,h); 512 threads.
// State columns S[:,v] are independent; split the k-dimension of each
// column across 4 adjacent lanes (tid = v*4 + seg) so cross-segment
// reductions are 2-step __shfl_xor inside one wave32. Each lane keeps
// 32 state values in registers -> per-step dependent chain is ~64 FMAs
// instead of ~256.
// =====================================================================
#define SEGS_ 4
#define KSEG_ (DK_ / SEGS_)   // 32

__global__ __launch_bounds__(512) void recurrence_kernel(
    const float* __restrict__ conv_out,   // [B,T,3072]  (q|k|v)
    const float* __restrict__ alpha,      // [B,T,H]
    const float* __restrict__ beta,       // [B,T,H]
    float* __restrict__ o)                // [B,T,H*DV]
{
    const int bh = blockIdx.x;            // 0..31
    const int b  = bh / H_;
    const int hh = bh % H_;
    const int tid = threadIdx.x;          // 0..511
    const int v   = tid >> 2;             // column 0..127
    const int seg = tid & (SEGS_ - 1);    // k-segment 0..3
    const int kbase = seg * KSEG_;

    __shared__ float ks[DK_];
    __shared__ float qs[DK_];

    float S[KSEG_];
#pragma unroll
    for (int i = 0; i < KSEG_; ++i) S[i] = 0.f;

    const float* basep = conv_out + (size_t)b * T_ * CONV_DIM_ + hh * DK_;
    const float* gatep = alpha + ((size_t)b * T_) * H_ + hh;
    const float* betap = beta  + ((size_t)b * T_) * H_ + hh;
    float* op = o + ((size_t)b * T_) * VAL_DIM_ + hh * DV_ + v;

    for (int t = 0; t < T_; ++t) {
        const float* rowp = basep + (size_t)t * CONV_DIM_;
        if (tid < DK_)            qs[tid]        = rowp[tid];
        else if (tid < 2 * DK_)   ks[tid - DK_]  = rowp[KEY_DIM_ + (tid - DK_)];
        const float vt = rowp[2 * KEY_DIM_ + v];
        const float al = gatep[(size_t)t * H_];
        const float be = betap[(size_t)t * H_];
        __syncthreads();

        float kv = 0.f;
#pragma unroll
        for (int i = 0; i < KSEG_; ++i) {
            float s = S[i] * al; S[i] = s;
            kv += ks[kbase + i] * s;
        }
        kv += __shfl_xor(kv, 1, 32);
        kv += __shfl_xor(kv, 2, 32);

        const float dv = (vt - kv) * be;

        float ov = 0.f;
#pragma unroll
        for (int i = 0; i < KSEG_; ++i) {
            float s = S[i] + ks[kbase + i] * dv; S[i] = s;
            ov += qs[kbase + i] * s;
        }
        ov += __shfl_xor(ov, 1, 32);
        ov += __shfl_xor(ov, 2, 32);

        if (seg == 0) op[(size_t)t * VAL_DIM_] = ov;
        __syncthreads();
    }
}

// =====================================================================
// Gated RMSNorm: o*rsqrt(mean(o^2)+eps)*norm_w * silu(z); 1 row / block
// =====================================================================
__global__ __launch_bounds__(128) void norm_gate_kernel(
    const float* __restrict__ o, const float* __restrict__ z,
    const float* __restrict__ norm_w, float* __restrict__ out)
{
    const int row = blockIdx.x;          // (b*T + t)*H + h
    const int d   = threadIdx.x;
    const int hh  = row % H_;
    const int bt  = row / H_;

    const float ov = o[(size_t)row * DV_ + d];
    float sq = ov * ov;
#pragma unroll
    for (int off = 16; off; off >>= 1) sq += __shfl_xor(sq, off, 32);
    __shared__ float part[4];
    if ((threadIdx.x & 31) == 0) part[threadIdx.x >> 5] = sq;
    __syncthreads();
    const float var = (part[0] + part[1] + part[2] + part[3]) * (1.f / 128.f);

    const float zz   = z[(size_t)bt * VAL_DIM_ + hh * DV_ + d];
    const float silu = zz / (1.f + __expf(-zz));
    out[(size_t)row * DV_ + d] = ov * rsqrtf(var + 1e-6f) * norm_w[d] * silu;
}

// =====================================================================
// host-side launcher
// =====================================================================
extern "C" void kernel_launch(void* const* d_in, const int* in_sizes, int n_in,
                              void* d_out, int out_size, void* d_ws, size_t ws_size,
                              hipStream_t stream)
{
    const float* hs      = (const float*)d_in[0];
    const float* qkv_w   = (const float*)d_in[1];
    const float* z_w     = (const float*)d_in[2];
    const float* b_w     = (const float*)d_in[3];
    const float* a_w     = (const float*)d_in[4];
    const float* conv_w  = (const float*)d_in[5];
    const float* dt_bias = (const float*)d_in[6];
    const float* A_log   = (const float*)d_in[7];
    const float* norm_w  = (const float*)d_in[8];
    const float* out_w   = (const float*)d_in[9];
    float* out = (float*)d_out;

    char* ws = (char*)d_ws;
    size_t off = 0;
    auto alloc = [&](size_t bytes) { char* p = ws + off; off += (bytes + 255) & ~size_t(255); return (float*)p; };

    float* mixed    = alloc((size_t)ROWS_ * CONV_DIM_ * 4);   // 48 MB
    float* conv_out = alloc((size_t)ROWS_ * CONV_DIM_ * 4);   // 48 MB
    float* zbuf     = alloc((size_t)ROWS_ * VAL_DIM_ * 4);    // 16 MB
    float* obuf     = alloc((size_t)ROWS_ * VAL_DIM_ * 4);    // 16 MB
    float* onorm    = alloc((size_t)ROWS_ * VAL_DIM_ * 4);    // 16 MB
    float* bb       = alloc((size_t)ROWS_ * H_ * 4);
    float* aa       = alloc((size_t)ROWS_ * H_ * 4);
    float* alpha    = alloc((size_t)ROWS_ * H_ * 4);
    float* beta     = alloc((size_t)ROWS_ * H_ * 4);

    // 1) mixed = hs @ qkv_w^T    [4096 x 3072]
    gemm_xwT_bf16<<<dim3(CONV_DIM_ / TILE_N, ROWS_ / TILE_M), 256, 0, stream>>>(
        hs, qkv_w, mixed, ROWS_, CONV_DIM_, HID_);

    // 2) z = hs @ z_w^T          [4096 x 1024]
    gemm_xwT_bf16<<<dim3(VAL_DIM_ / TILE_N, ROWS_ / TILE_M), 256, 0, stream>>>(
        hs, z_w, zbuf, ROWS_, VAL_DIM_, HID_);

    // 3) bb, aa small projections
    {
        int waves = ROWS_ * H_ * 2;
        int threads = waves * 32;
        ba_proj_kernel<<<(threads + 255) / 256, 256, 0, stream>>>(hs, b_w, a_w, bb, aa);
    }

    // 4) causal depthwise conv
    {
        int total = ROWS_ * CONV_DIM_;
        conv_kernel<<<(total + 255) / 256, 256, 0, stream>>>(mixed, conv_w, conv_out, total);
    }

    // 5) alpha/beta gates
    {
        int total = ROWS_ * H_;
        gate_kernel<<<(total + 255) / 256, 256, 0, stream>>>(bb, aa, A_log, dt_bias, alpha, beta, total);
    }

    // 6) gated delta-rule scan (k-dim split across 4 lanes per column)
    recurrence_kernel<<<B_ * H_, 512, 0, stream>>>(conv_out, alpha, beta, obuf);

    // 7) gated RMSNorm
    norm_gate_kernel<<<ROWS_ * H_, 128, 0, stream>>>(obuf, zbuf, norm_w, onorm);

    // 8) out = onorm @ out_w^T   [4096 x 1024]
    gemm_xwT_bf16<<<dim3(HID_ / TILE_N, ROWS_ / TILE_M), 256, 0, stream>>>(
        onorm, out_w, out, ROWS_, HID_, VAL_DIM_);
}
